// N3SAGE_6098853560426
// MI455X (gfx1250) — matrix operand order repaired and checked by reading, not compiled, not run on hardware
//
#include <hip/hip_runtime.h>

#define NN 16000
#define EE 128000

typedef __attribute__((ext_vector_type(16))) __bf16 v16bf;
typedef __attribute__((ext_vector_type(8)))  float  v8f;

union FragB16 { uint4 q[2]; v16bf v; };

__device__ __forceinline__ unsigned short f2bf(float f) {
    unsigned int u = __float_as_uint(f);
    u += 0x7FFFu + ((u >> 16) & 1u);          // round-to-nearest-even
    return (unsigned short)(u >> 16);
}
__device__ __forceinline__ float bf2f(unsigned short h) {
    return __uint_as_float(((unsigned int)h) << 16);
}

// async global->LDS 16B copy (gfx1250 GLOBAL_LOAD_ASYNC_TO_LDS_B128, ASYNCcnt)
__device__ __forceinline__ void async_cp_b128(unsigned lds_addr, const unsigned int* gaddr) {
    asm volatile("global_load_async_to_lds_b128 %0, %1, off"
                 :: "v"(lds_addr), "v"(gaddr) : "memory");
}
__device__ __forceinline__ void wait_async0() {
    asm volatile("s_wait_asynccnt 0x0" ::: "memory");
}

// ---------------- small utility kernels ----------------

__global__ void k_cast_bf16(const float* __restrict__ in, unsigned short* __restrict__ out, int n) {
    int i = blockIdx.x * blockDim.x + threadIdx.x;
    if (i < n) out[i] = f2bf(in[i]);
}

// Pack f32 weight [Fin x Fout] row-major into WMMA B-fragment order:
// Wp[((ko*(Fout/16)+nt)*32 + lane)*8 + v] = {bf16 W[k0,n], bf16 W[k0+1,n]}
//   n = nt*16 + (lane&15),  k0 = ko*32 + (v>>2)*16 + (lane>>4)*8 + 2*(v&3)
__global__ void k_pack_w(const float* __restrict__ W, unsigned int* __restrict__ Wp,
                         int Fin, int Fout) {
    int i = blockIdx.x * blockDim.x + threadIdx.x;
    int total = (Fin * Fout) >> 1;
    if (i >= total) return;
    int v    = i & 7;
    int lane = (i >> 3) & 31;
    int rest = i >> 8;
    int ntiles = Fout >> 4;
    int nt = rest % ntiles;
    int ko = rest / ntiles;
    int n  = nt * 16 + (lane & 15);
    int k0 = ko * 32 + (v >> 2) * 16 + (lane >> 4) * 8 + 2 * (v & 3);
    unsigned short lo = f2bf(W[(size_t)k0 * Fout + n]);
    unsigned short hi = f2bf(W[(size_t)(k0 + 1) * Fout + n]);
    Wp[i] = (unsigned int)lo | ((unsigned int)hi << 16);
}

__global__ void k_degree(const int* __restrict__ dst, float* __restrict__ deg, int n) {
    int i = blockIdx.x * blockDim.x + threadIdx.x;
    if (i < n) atomicAdd(&deg[dst[i]], 1.0f);
}

__global__ void k_invdeg(float* __restrict__ deg, int n) {
    int i = blockIdx.x * blockDim.x + threadIdx.x;
    if (i < n) deg[i] = 1.0f / fmaxf(deg[i], 1.0f);
}

// one block per edge: agg[dst] += x[src]  (f32 source)
__global__ void k_scatter_f32(const float* __restrict__ x, const int* __restrict__ src,
                              const int* __restrict__ dst, float* __restrict__ agg, int F) {
    const int e = blockIdx.x;
    const int s = src[e], d = dst[e];
    const float* xr = x + (size_t)s * F;
    float* ar = agg + (size_t)d * F;
    for (int f = threadIdx.x; f < F; f += blockDim.x)
        atomicAdd(&ar[f], xr[f]);
}

// one block per edge: agg[dst] += x[src]  (bf16 source)
__global__ void k_scatter_bf16(const unsigned short* __restrict__ x, const int* __restrict__ src,
                               const int* __restrict__ dst, float* __restrict__ agg, int F) {
    const int e = blockIdx.x;
    const int s = src[e], d = dst[e];
    const unsigned short* xr = x + (size_t)s * F;
    float* ar = agg + (size_t)d * F;
    for (int f = threadIdx.x; f < F; f += blockDim.x)
        atomicAdd(&ar[f], bf2f(xr[f]));
}

// mean = agg * invdeg[row], packed to bf16 row-major (natural A-fragment source)
__global__ void k_meanpack(const float* __restrict__ agg, const float* __restrict__ invdeg,
                           unsigned short* __restrict__ out, int total, int shift) {
    int i = blockIdx.x * blockDim.x + threadIdx.x;
    if (i < total) out[i] = f2bf(agg[i] * invdeg[i >> shift]);
}

// ---------------- fused SAGE GEMM: out = mean@Wl + x@Wr + b (+ReLU) ----------------
// block = 128 threads (4 waves). Block tile: 128 rows x 64 cols.
// Each wave: 2 M-tiles x 4 N-tiles (32x64), 8 f32 accumulators.
// All 4 waves share the same 64-column group, so both packed-B K-tiles (Wl, Wr;
// 4KB contiguous each) are staged once per block into LDS with async DMA copies,
// double-buffered, and fragments are read back via ds_load_b128.
__global__ __launch_bounds__(128)
void k_gemm_sage(const unsigned short* __restrict__ Amean,
                 const unsigned short* __restrict__ Ax,
                 const unsigned int*  __restrict__ Wpl,
                 const unsigned int*  __restrict__ Wpr,
                 const float* __restrict__ bias,
                 int Fin, int Fout,
                 unsigned short* __restrict__ outB,
                 float* __restrict__ outF,
                 int relu)
{
    __shared__ unsigned int lbuf[2][2048];    // [buf][{Wl:1024, Wr:1024} uints] = 16KB

    const int tid  = threadIdx.x;
    const int wave = tid >> 5;
    const int lane = tid & 31;
    const int lm   = lane & 15;               // row (A) / col (B,C) within tile
    const int lh   = lane >> 4;               // half-select
    const int ntiles = Fout >> 4;
    const int ntg    = blockIdx.y;            // 64-column group
    const int rowBase = blockIdx.x * 128 + wave * 32;
    const int kTiles = Fin >> 5;

    const unsigned int* aM0 = (const unsigned int*)Amean + (size_t)(rowBase + lm)      * (Fin >> 1);
    const unsigned int* aM1 = (const unsigned int*)Amean + (size_t)(rowBase + 16 + lm) * (Fin >> 1);
    const unsigned int* aX0 = (const unsigned int*)Ax    + (size_t)(rowBase + lm)      * (Fin >> 1);
    const unsigned int* aX1 = (const unsigned int*)Ax    + (size_t)(rowBase + 16 + lm) * (Fin >> 1);

    // stage packed B (Wl+Wr) for K-tile ko into LDS buffer b: 8KB via 4x b128/thread
    auto stage = [&](int ko, int b) {
        const unsigned int* gl = Wpl + (size_t)(ko * ntiles + ntg * 4) * 256 + tid * 4;
        const unsigned int* gr = Wpr + (size_t)(ko * ntiles + ntg * 4) * 256 + tid * 4;
        unsigned d = (unsigned)(size_t)(const void*)&lbuf[b][0] + (unsigned)tid * 16u;
        async_cp_b128(d,          gl);
        async_cp_b128(d + 2048u,  gl + 512);
        async_cp_b128(d + 4096u,  gr);
        async_cp_b128(d + 6144u,  gr + 512);
    };

    stage(0, 0);

    v8f acc[2][4] = {};
    int cb = 0;
    for (int ko = 0; ko < kTiles; ++ko) {
        wait_async0();          // my async writes for buffer cb have landed in LDS
        __syncthreads();        // everyone's writes visible; prior reads of cb^1 done
        if (ko + 1 < kTiles) stage(ko + 1, cb ^ 1);

        // A fragments per ISA 16-bit 16x32 layout:
        // lanes 0-15: uints {0..3, 8..11} of the 16-uint K-tile; lanes 16-31: {4..7, 12..15}
        const int abase = ko * 16 + lh * 4;
        FragB16 am[2], ax[2];
        am[0].q[0] = *(const uint4*)(aM0 + abase);
        am[0].q[1] = *(const uint4*)(aM0 + abase + 8);
        am[1].q[0] = *(const uint4*)(aM1 + abase);
        am[1].q[1] = *(const uint4*)(aM1 + abase + 8);
        ax[0].q[0] = *(const uint4*)(aX0 + abase);
        ax[0].q[1] = *(const uint4*)(aX0 + abase + 8);
        ax[1].q[0] = *(const uint4*)(aX1 + abase);
        ax[1].q[1] = *(const uint4*)(aX1 + abase + 8);
        __builtin_prefetch(aM0 + abase + 16, 0, 1);
        __builtin_prefetch(aM1 + abase + 16, 0, 1);
        __builtin_prefetch(aX0 + abase + 16, 0, 1);
        __builtin_prefetch(aX1 + abase + 16, 0, 1);

        const unsigned int* lb = &lbuf[cb][0];
#pragma unroll
        for (int j = 0; j < 4; ++j) {
            FragB16 bl, br;
            bl.q[0] = *(const uint4*)(lb + j * 256 + lane * 8);
            bl.q[1] = *(const uint4*)(lb + j * 256 + lane * 8 + 4);
            br.q[0] = *(const uint4*)(lb + 1024 + j * 256 + lane * 8);
            br.q[1] = *(const uint4*)(lb + 1024 + j * 256 + lane * 8 + 4);
#pragma unroll
            for (int mi = 0; mi < 2; ++mi) {
                acc[mi][j] = __builtin_amdgcn_wmma_f32_16x16x32_bf16(
                    false, am[mi].v, false, bl.v, (short)0, acc[mi][j], false, false);
                acc[mi][j] = __builtin_amdgcn_wmma_f32_16x16x32_bf16(
                    false, ax[mi].v, false, br.v, (short)0, acc[mi][j], false, false);
            }
        }
        cb ^= 1;
    }

    // C/D layout: lanes 0-15 -> M=r, lanes 16-31 -> M=8+r; N = lm
    const int colBase = ntg * 64 + lm;
#pragma unroll
    for (int mi = 0; mi < 2; ++mi) {
#pragma unroll
        for (int j = 0; j < 4; ++j) {
            const int col = colBase + j * 16;
            const float bb = bias[col];
#pragma unroll
            for (int r = 0; r < 8; ++r) {
                const int orow = rowBase + mi * 16 + lh * 8 + r;
                float v = acc[mi][j][r] + bb;
                if (relu) v = fmaxf(v, 0.0f);
                if (outF) outF[(size_t)orow * Fout + col] = v;
                else      outB[(size_t)orow * Fout + col] = f2bf(v);
            }
        }
    }
}

// ---------------- log-softmax over 64 columns, one wave per row ----------------
__global__ __launch_bounds__(256)
void k_logsoftmax(float* __restrict__ out, int n) {
    const int wave = threadIdx.x >> 5;
    const int lane = threadIdx.x & 31;
    const int row = blockIdx.x * 8 + wave;
    if (row >= n) return;
    float* p = out + (size_t)row * 64;
    float a = p[lane], b = p[lane + 32];
    float m = fmaxf(a, b);
#pragma unroll
    for (int o = 16; o > 0; o >>= 1) m = fmaxf(m, __shfl_xor(m, o, 32));
    float s = expf(a - m) + expf(b - m);
#pragma unroll
    for (int o = 16; o > 0; o >>= 1) s += __shfl_xor(s, o, 32);
    float l = m + logf(s);
    p[lane]      = a - l;
    p[lane + 32] = b - l;
}

// ---------------- launch ----------------

extern "C" void kernel_launch(void* const* d_in, const int* in_sizes, int n_in,
                              void* d_out, int out_size, void* d_ws, size_t ws_size,
                              hipStream_t stream) {
    const float* x   = (const float*)d_in[0];
    const int*   ei  = (const int*)  d_in[1];
    const float* W1l = (const float*)d_in[2];
    const float* b1  = (const float*)d_in[3];
    const float* W1r = (const float*)d_in[4];
    const float* W2l = (const float*)d_in[5];
    const float* b2  = (const float*)d_in[6];
    const float* W2r = (const float*)d_in[7];
    const float* W3l = (const float*)d_in[8];
    const float* b3  = (const float*)d_in[9];
    const float* W3r = (const float*)d_in[10];
    const int* src = ei;
    const int* dst = ei + EE;
    float* out = (float*)d_out;

    char* p = (char*)d_ws;
    auto wsAlloc = [&](size_t bytes) -> void* {
        void* r = (void*)p;
        p += (bytes + 255) & ~(size_t)255;
        return r;
    };
    unsigned short* xB    = (unsigned short*)wsAlloc((size_t)NN * 512 * 2);
    unsigned int*   wp1l  = (unsigned int*)  wsAlloc((size_t)512 * 4096 * 2);
    unsigned int*   wp1r  = (unsigned int*)  wsAlloc((size_t)512 * 4096 * 2);
    unsigned int*   wp2l  = (unsigned int*)  wsAlloc((size_t)4096 * 1024 * 2);
    unsigned int*   wp2r  = (unsigned int*)  wsAlloc((size_t)4096 * 1024 * 2);
    unsigned int*   wp3l  = (unsigned int*)  wsAlloc((size_t)1024 * 64 * 2);
    unsigned int*   wp3r  = (unsigned int*)  wsAlloc((size_t)1024 * 64 * 2);
    float*          deg   = (float*)         wsAlloc((size_t)NN * 4);
    float*          agg   = (float*)         wsAlloc((size_t)NN * 4096 * 4);
    unsigned short* meanB = (unsigned short*)wsAlloc((size_t)NN * 4096 * 2);
    unsigned short* h1    = (unsigned short*)wsAlloc((size_t)NN * 4096 * 2);
    unsigned short* h2    = (unsigned short*)wsAlloc((size_t)NN * 1024 * 2);

    // conversions / packing (per launch, deterministic)
    k_cast_bf16<<<(NN * 512 + 255) / 256, 256, 0, stream>>>(x, xB, NN * 512);
    k_pack_w<<<((512 * 4096 / 2) + 255) / 256, 256, 0, stream>>>(W1l, wp1l, 512, 4096);
    k_pack_w<<<((512 * 4096 / 2) + 255) / 256, 256, 0, stream>>>(W1r, wp1r, 512, 4096);
    k_pack_w<<<((4096 * 1024 / 2) + 255) / 256, 256, 0, stream>>>(W2l, wp2l, 4096, 1024);
    k_pack_w<<<((4096 * 1024 / 2) + 255) / 256, 256, 0, stream>>>(W2r, wp2r, 4096, 1024);
    k_pack_w<<<((1024 * 64 / 2) + 255) / 256, 256, 0, stream>>>(W3l, wp3l, 1024, 64);
    k_pack_w<<<((1024 * 64 / 2) + 255) / 256, 256, 0, stream>>>(W3r, wp3r, 1024, 64);

    hipMemsetAsync(deg, 0, (size_t)NN * 4, stream);
    k_degree<<<(EE + 255) / 256, 256, 0, stream>>>(dst, deg, EE);
    k_invdeg<<<(NN + 255) / 256, 256, 0, stream>>>(deg, NN);

    // ---- layer 1: 512 -> 4096, ReLU ----
    hipMemsetAsync(agg, 0, (size_t)NN * 512 * 4, stream);
    k_scatter_f32<<<EE, 256, 0, stream>>>(x, src, dst, agg, 512);
    k_meanpack<<<(NN * 512 + 255) / 256, 256, 0, stream>>>(agg, deg, meanB, NN * 512, 9);
    {
        dim3 g(NN / 128, 4096 / 64);
        k_gemm_sage<<<g, 128, 0, stream>>>(meanB, xB, wp1l, wp1r, b1, 512, 4096,
                                           h1, nullptr, 1);
    }

    // ---- layer 2: 4096 -> 1024, ReLU ----
    hipMemsetAsync(agg, 0, (size_t)NN * 4096 * 4, stream);
    k_scatter_bf16<<<EE, 256, 0, stream>>>(h1, src, dst, agg, 4096);
    k_meanpack<<<(NN * 4096 + 255) / 256, 256, 0, stream>>>(agg, deg, meanB, NN * 4096, 12);
    {
        dim3 g(NN / 128, 1024 / 64);
        k_gemm_sage<<<g, 128, 0, stream>>>(meanB, h1, wp2l, wp2r, b2, 4096, 1024,
                                           h2, nullptr, 1);
    }

    // ---- layer 3: 1024 -> 64, f32 out ----
    hipMemsetAsync(agg, 0, (size_t)NN * 1024 * 4, stream);
    k_scatter_bf16<<<EE, 256, 0, stream>>>(h2, src, dst, agg, 1024);
    k_meanpack<<<(NN * 1024 + 255) / 256, 256, 0, stream>>>(agg, deg, meanB, NN * 1024, 10);
    {
        dim3 g(NN / 128, 1);
        k_gemm_sage<<<g, 128, 0, stream>>>(meanB, h2, wp3l, wp3r, b3, 1024, 64,
                                           nullptr, out, 0);
    }

    k_logsoftmax<<<NN / 8, 256, 0, stream>>>(out, NN);
}
